// EpiGraphModelV3_6382321401961
// MI455X (gfx1250) — matrix operand order repaired and checked by reading, not compiled, not run on hardware
//
#include <hip/hip_runtime.h>
#include <hip/hip_bf16.h>
#include <stdint.h>

typedef __attribute__((ext_vector_type(16))) __bf16 v16bf;
typedef __attribute__((ext_vector_type(8)))  float  v8f;

// ---------------------------------------------------------------------------
// helpers
// ---------------------------------------------------------------------------
__device__ __forceinline__ unsigned short f2bf_bits(float f) {
  uint32_t u = __builtin_bit_cast(uint32_t, f);
  u += 0x7FFFu + ((u >> 16) & 1u);          // round-to-nearest-even
  return (unsigned short)(u >> 16);
}
__device__ __forceinline__ __bf16 f2bf(float f) {
  return __builtin_bit_cast(__bf16, f2bf_bits(f));
}
__device__ __forceinline__ float sigf(float x) { return 1.0f / (1.0f + __expf(-x)); }

union Frag16 { uint4 u[2]; v16bf v; };

// ---------------------------------------------------------------------------
// prep: activations fp32 [M,K] -> bf16 [M,Kpad], Kpad = 1<<kshift, zero pad
// ---------------------------------------------------------------------------
__global__ __launch_bounds__(256) void prep_act_kernel(
    const float* __restrict__ in, unsigned short* __restrict__ out,
    int M, int K, int kshift)
{
  const int Kpad = 1 << kshift;
  size_t total = (size_t)M << kshift;
  size_t stride = (size_t)gridDim.x * blockDim.x;
  for (size_t idx = (size_t)blockIdx.x * blockDim.x + threadIdx.x;
       idx < total; idx += stride) {
    int k = (int)(idx & (Kpad - 1));
    size_t row = idx >> kshift;
    float v = (k < K) ? in[row * K + k] : 0.f;
    out[idx] = f2bf_bits(v);
  }
}

// ---------------------------------------------------------------------------
// prep: weights -> bf16 [N,Kpad].  trans==0: src is [K,N]; trans==1: src [N,K]
// ---------------------------------------------------------------------------
__global__ __launch_bounds__(256) void prep_w_kernel(
    const float* __restrict__ W, unsigned short* __restrict__ out,
    int N, int K, int kshift, int trans)
{
  const int Kpad = 1 << kshift;
  size_t total = (size_t)N << kshift;
  size_t stride = (size_t)gridDim.x * blockDim.x;
  for (size_t idx = (size_t)blockIdx.x * blockDim.x + threadIdx.x;
       idx < total; idx += stride) {
    int k = (int)(idx & (Kpad - 1));
    int n = (int)(idx >> kshift);
    float v = 0.f;
    if (k < K) v = trans ? W[(size_t)n * K + k] : W[(size_t)k * N + n];
    out[idx] = f2bf_bits(v);
  }
}

// ---------------------------------------------------------------------------
// bf16 WMMA GEMM (branchless, vectorized):
//   C[M,N] = Abf[M,Kpad] @ Bbf[N,Kpad]^T + b1 + b2
// One 16x16 C tile per wave; per k-chunk: 4x global_load_b128 + 1x v_wmma.
// ---------------------------------------------------------------------------
__global__ __launch_bounds__(128) void gemm_bf16_kernel(
    const unsigned short* __restrict__ Ab,
    const unsigned short* __restrict__ Bb, int Kpad,
    const float* __restrict__ b1, const float* __restrict__ b2,
    float* __restrict__ C, int ldc, int M, int N)
{
  const int tid  = threadIdx.x;
  const int lane = tid & 31;
  const int wv   = tid >> 5;
  const int gw   = blockIdx.x * 4 + wv;      // global wave id -> tile id
  const int ntiles = N >> 4;
  const int mt = gw / ntiles;
  const int nt = gw % ntiles;
  if (mt * 16 >= M) return;

  const int col = lane & 15;                 // lane%16
  const int hi  = lane >> 4;                 // lane/16
  const int arow = mt * 16 + col;
  const int ncol = nt * 16 + col;

  const uint4* ap = (const uint4*)(Ab + (size_t)arow * Kpad);
  const uint4* bp = (const uint4*)(Bb + (size_t)ncol * Kpad);

  v8f acc = {0.f, 0.f, 0.f, 0.f, 0.f, 0.f, 0.f, 0.f};

  const int KC = Kpad >> 5;
  #pragma unroll 2
  for (int kc = 0; kc < KC; ++kc) {
    Frag16 a, b;
    a.u[0] = ap[kc * 4 + hi];                // 8 bf16 @ kc*32 + hi*8
    a.u[1] = ap[kc * 4 + hi + 2];            // 8 bf16 @ kc*32 + hi*8 + 16
    b.u[0] = bp[kc * 4 + hi];                // 8 bf16 @ kc*32 + hi*8
    b.u[1] = bp[kc * 4 + hi + 2];            // 8 bf16 @ kc*32 + hi*8 + 16
    acc = __builtin_amdgcn_wmma_f32_16x16x32_bf16(
        false, a.v, false, b.v, (short)0, acc, false, false);
  }

  float bb = 0.f;
  if (b1) bb += b1[ncol];
  if (b2) bb += b2[ncol];

  #pragma unroll
  for (int r = 0; r < 8; ++r) {
    int m = mt * 16 + r + (hi << 3);         // D row = r + 8*(lane/16)
    C[(size_t)m * ldc + ncol] = acc[r] + bb;
  }
}

// ---------------------------------------------------------------------------
// GATv2 attention + aggregate + bias + ELU.
// One block per graph (G = b*t = 16384), blockDim = H*128.
// ---------------------------------------------------------------------------
__global__ void gat_attn_kernel(
    const float* __restrict__ Hl, const float* __restrict__ Hr,
    const float* __restrict__ att, const float* __restrict__ bias,
    const int* __restrict__ edge, float* __restrict__ out, int H)
{
  const int g   = blockIdx.x;
  const int tid = threadIdx.x;
  const int h   = tid >> 7;
  const int d   = tid & 127;
  const int lane = tid & 31;
  const int wave = tid >> 5;
  const int C   = H * 128;

  const float* hl = Hl + (size_t)g * 5 * C;
  const float* hr = Hr + (size_t)g * 5 * C;

  unsigned mask = 0;
  #pragma unroll
  for (int e = 0; e < 12; ++e) {
    int s = edge[e], dst = edge[12 + e];
    mask |= 1u << (dst * 5 + s);
  }
  #pragma unroll
  for (int i = 0; i < 5; ++i) mask |= 1u << (i * 5 + i);

  __shared__ float sc[5][5][2];
  __shared__ float red[8];

  const float aw = att[h * 128 + d];

  for (int i = 0; i < 5; ++i) {
    for (int j = 0; j < 5; ++j) {
      float x = hr[i * C + h * 128 + d] + hl[j * C + h * 128 + d];
      x = (x > 0.f) ? x : 0.2f * x;          // leaky_relu(0.2)
      float v = x * aw;
      v += __shfl_xor(v, 16);
      v += __shfl_xor(v, 8);
      v += __shfl_xor(v, 4);
      v += __shfl_xor(v, 2);
      v += __shfl_xor(v, 1);
      if (lane == 0) red[wave] = v;
      __syncthreads();
      if (tid == 0) {
        for (int hh = 0; hh < H; ++hh)
          sc[i][j][hh] = red[hh * 4] + red[hh * 4 + 1] +
                         red[hh * 4 + 2] + red[hh * 4 + 3];
      }
      __syncthreads();
    }
  }

  for (int i = 0; i < 5; ++i) {
    float s[5];
    float mx = -1e30f;
    #pragma unroll
    for (int j = 0; j < 5; ++j) {
      s[j] = ((mask >> (i * 5 + j)) & 1u) ? sc[i][j][h] : -1e30f;
      mx = fmaxf(mx, s[j]);
    }
    float den = 0.f;
    #pragma unroll
    for (int j = 0; j < 5; ++j) { s[j] = __expf(s[j] - mx); den += s[j]; }
    float o = 0.f;
    #pragma unroll
    for (int j = 0; j < 5; ++j) o += s[j] * hl[j * C + h * 128 + d];
    o = o / den + bias[h * 128 + d];
    o = (o > 0.f) ? o : (__expf(o) - 1.f);   // ELU
    out[(size_t)g * 5 * C + i * C + h * 128 + d] = o;
  }
}

// ---------------------------------------------------------------------------
// LSTM recurrence, weight-resident WMMA + software-pipelined Xp loads.
// Xp : [640*128, 512] row = q*128+t  (x@Wih^T + bih + bhh, precomputed)
// Whh: [512,128]; out: [640*128, 128] row = q*128+t
// Wave w owns gate columns [16w,16w+16) of all four gates, so i/f/g/o of a
// cell live in one lane's accumulators. Whh^T B-frags persist in VGPRs.
// Next step's Xp slice is loaded into registers while this step computes.
// ---------------------------------------------------------------------------
__global__ __launch_bounds__(256) void lstm_recurrent_kernel(
    const float* __restrict__ Xp, const float* __restrict__ Whh,
    float* __restrict__ out)
{
  __shared__ unsigned short sHh[16 * 128];   // h tile, bf16
  const unsigned* sH32 = (const unsigned*)sHh;

  const int tid  = threadIdx.x;
  const int w    = tid >> 5;                 // wave -> gate column block
  const int lane = tid & 31;
  const int col  = lane & 15;
  const int hi   = lane >> 4;
  const int q0   = blockIdx.x * 16;

  // resident B-frags: B[k][n] = Whh[gate*128 + 16w + n][k], contiguous in k
  v16bf B[4][4];
  #pragma unroll
  for (int gg = 0; gg < 4; ++gg) {
    const float* wr = Whh + (size_t)(gg * 128 + w * 16 + col) * 128;
    #pragma unroll
    for (int kc = 0; kc < 4; ++kc) {
      v16bf f;
      #pragma unroll
      for (int e = 0; e < 16; ++e) f[e] = f2bf(wr[kc * 32 + hi * 16 + e]);
      B[gg][kc] = f;
    }
  }

  for (int p = tid; p < 16 * 128; p += 256) sHh[p] = 0;  // h0 = 0
  float c[8];
  #pragma unroll
  for (int r = 0; r < 8; ++r) c[r] = 0.f;                // c0 = 0

  // prefetch Xp slice for t = 0
  v8f xn[4];
  #pragma unroll
  for (int gg = 0; gg < 4; ++gg)
    #pragma unroll
    for (int r = 0; r < 8; ++r) {
      int m = r + (hi << 3);
      xn[gg][r] = Xp[((size_t)(q0 + m) * 128 + 0) * 512 +
                     gg * 128 + w * 16 + col];
    }
  __syncthreads();

  for (int t = 0; t < 128; ++t) {
    // consume prefetched slice, immediately issue loads for t+1
    v8f acc[4];
    #pragma unroll
    for (int gg = 0; gg < 4; ++gg) acc[gg] = xn[gg];

    const int tn = (t < 127) ? (t + 1) : 127;
    #pragma unroll
    for (int gg = 0; gg < 4; ++gg)
      #pragma unroll
      for (int r = 0; r < 8; ++r) {
        int m = r + (hi << 3);
        xn[gg][r] = Xp[((size_t)(q0 + m) * 128 + tn) * 512 +
                       gg * 128 + w * 16 + col];
      }

    // gates += h @ Whh^T
    #pragma unroll
    for (int kc = 0; kc < 4; ++kc) {
      v16bf a;
      #pragma unroll
      for (int e = 0; e < 16; e += 2) {
        int k = kc * 32 + (hi << 4) + e;
        unsigned pr = sH32[(col * 128 + k) >> 1];
        a[e]     = __builtin_bit_cast(__bf16, (unsigned short)(pr & 0xFFFFu));
        a[e + 1] = __builtin_bit_cast(__bf16, (unsigned short)(pr >> 16));
      }
      #pragma unroll
      for (int gg = 0; gg < 4; ++gg)
        acc[gg] = __builtin_amdgcn_wmma_f32_16x16x32_bf16(
            false, a, false, B[gg][kc], (short)0, acc[gg], false, false);
    }
    __syncthreads();                         // all sH reads done

    #pragma unroll
    for (int r = 0; r < 8; ++r) {
      float gi = sigf(acc[0][r]);
      float gf = sigf(acc[1][r]);
      float gc = tanhf(acc[2][r]);
      float go = sigf(acc[3][r]);
      c[r] = gf * c[r] + gi * gc;
      float hn = go * tanhf(c[r]);
      int m = r + (hi << 3);
      int j = w * 16 + col;
      out[((size_t)(q0 + m) * 128 + t) * 128 + j] = hn;
      sHh[m * 128 + j] = f2bf_bits(hn);
    }
    __syncthreads();                         // all sH writes done
  }
}

// ---------------------------------------------------------------------------
// Tail: layernorm + temporal attention + skip MLP + final FC.
// ---------------------------------------------------------------------------
__global__ __launch_bounds__(128) void tail_kernel(
    const float* __restrict__ out1, const float* __restrict__ x,
    const float* __restrict__ lng, const float* __restrict__ lnb,
    const float* __restrict__ taW, const float* __restrict__ taB,
    const float* __restrict__ skW1, const float* __restrict__ skB1,
    const float* __restrict__ skW2, const float* __restrict__ skB2,
    const float* __restrict__ fcW1, const float* __restrict__ fcB1,
    const float* __restrict__ fcW2, const float* __restrict__ fcB2,
    float* __restrict__ y)
{
  const int q = blockIdx.x;
  const int t = threadIdx.x;

  __shared__ float sMu[128], sRs[128], sAttn[128], sTmp[128];
  __shared__ float sCtx[160], sS1[64], sF1[64];

  const float* row = out1 + ((size_t)q * 128 + t) * 128;
  float sum = 0.f, sq = 0.f;
  for (int j = 0; j < 128; ++j) { float v = row[j]; sum += v; sq += v * v; }
  float mu = sum * (1.f / 128.f);
  float var = sq * (1.f / 128.f) - mu * mu;
  float rs = rsqrtf(var + 1e-5f);
  sMu[t] = mu; sRs[t] = rs;

  float s = taB[0];
  for (int j = 0; j < 128; ++j) {
    float v = (row[j] - mu) * rs * lng[j] + lnb[j];
    s += v * taW[j];
  }
  sTmp[t] = s;
  __syncthreads();

  float mx = -1e30f;
  for (int u = 0; u < 128; ++u) mx = fmaxf(mx, sTmp[u]);
  float e = __expf(s - mx);
  sAttn[t] = e;
  __syncthreads();
  float den = 0.f;
  for (int u = 0; u < 128; ++u) den += sAttn[u];

  float ctx = 0.f;
  for (int u = 0; u < 128; ++u) {
    float v = (out1[((size_t)q * 128 + u) * 128 + t] - sMu[u]) * sRs[u] * lng[t]
              + lnb[t];
    ctx += sAttn[u] * v;
  }
  sCtx[t] = ctx / den;
  __syncthreads();

  const int b_ = q / 5, n_ = q % 5;
  const float* xin = x + (((size_t)b_ * 128 + 127) * 5 + n_) * 29;
  if (t < 64) {
    float a = skB1[t];
    for (int k = 0; k < 13; ++k) a += xin[k] * skW1[k * 64 + t];
    sS1[t] = (a > 0.f) ? a : 0.f;
  }
  __syncthreads();
  if (t < 32) {
    float a = skB2[t];
    for (int k = 0; k < 64; ++k) a += sS1[k] * skW2[k * 32 + t];
    sCtx[128 + t] = a;
  }
  __syncthreads();
  if (t < 64) {
    float a = fcB1[t];
    for (int k = 0; k < 160; ++k) a += sCtx[k] * fcW1[k * 64 + t];
    sF1[t] = (a > 0.f) ? a : 0.f;
  }
  __syncthreads();
  if (t == 0) {
    float a = fcB2[0];
    for (int k = 0; k < 64; ++k) a += sF1[k] * fcW2[k];
    y[q] = a;
  }
}

// ---------------------------------------------------------------------------
// launcher
// ---------------------------------------------------------------------------
extern "C" void kernel_launch(void* const* d_in, const int* in_sizes, int n_in,
                              void* d_out, int out_size, void* d_ws, size_t ws_size,
                              hipStream_t stream) {
  (void)in_sizes; (void)n_in; (void)out_size; (void)ws_size;

  const float* x       = (const float*)d_in[0];
  const int*   edge    = (const int*)d_in[1];
  const float* g1_Wl   = (const float*)d_in[2];
  const float* g1_bl   = (const float*)d_in[3];
  const float* g1_Wr   = (const float*)d_in[4];
  const float* g1_br   = (const float*)d_in[5];
  const float* g1_att  = (const float*)d_in[6];
  const float* g1_bias = (const float*)d_in[7];
  const float* g2_Wl   = (const float*)d_in[8];
  const float* g2_bl   = (const float*)d_in[9];
  const float* g2_Wr   = (const float*)d_in[10];
  const float* g2_br   = (const float*)d_in[11];
  const float* g2_att  = (const float*)d_in[12];
  const float* g2_bias = (const float*)d_in[13];
  const float* Wih0    = (const float*)d_in[14];
  const float* Whh0    = (const float*)d_in[15];
  const float* bih0    = (const float*)d_in[16];
  const float* bhh0    = (const float*)d_in[17];
  const float* Wih1    = (const float*)d_in[18];
  const float* Whh1    = (const float*)d_in[19];
  const float* bih1    = (const float*)d_in[20];
  const float* bhh1    = (const float*)d_in[21];
  const float* ln_g    = (const float*)d_in[22];
  const float* ln_b    = (const float*)d_in[23];
  const float* ta_W    = (const float*)d_in[24];
  const float* ta_b    = (const float*)d_in[25];
  const float* sk_W1   = (const float*)d_in[26];
  const float* sk_b1   = (const float*)d_in[27];
  const float* sk_W2   = (const float*)d_in[28];
  const float* sk_b2   = (const float*)d_in[29];
  const float* fc_W1   = (const float*)d_in[30];
  const float* fc_b1   = (const float*)d_in[31];
  const float* fc_W2   = (const float*)d_in[32];
  const float* fc_b2   = (const float*)d_in[33];
  float* y = (float*)d_out;                       // [128,5,1] = 640 floats

  // workspace layout (float units)
  const size_t M = 81920;
  float* W0 = (float*)d_ws;                       // 41,943,040 f  (Xp)
  float* W1 = W0 + (size_t)41943040;              // 20,971,520 f
  float* W2 = W1 + (size_t)20971520;              // 20,971,520 f
  float* W3 = W2 + (size_t)20971520;              // 20,971,520 f
  unsigned short* ABF = (unsigned short*)(W3 + (size_t)20971520); // 20,971,520 bf16
  unsigned short* WB1 = ABF + (size_t)20971520;   // 65,536 bf16
  unsigned short* WB2 = WB1 + (size_t)65536;      // 65,536 bf16

  float* Hl1  = W1;                               // [81920,256]
  float* Hr1  = W2;
  float* h1   = W3;
  float* Hl2  = W1;                               // [81920,128]
  float* Hr2  = W1 + (size_t)10485760;
  float* h2   = W2;                               // LSTM input alias via .view
  float* Xp0  = W0;                               // [81920,512]
  float* out0 = W2 + (size_t)10485760;
  float* Xp1  = W0;
  float* outL = W3;

  dim3 blk128(128), blk256(256);

  // ---- GATv2 layer 1: x[81920,29] -> Hl1/Hr1 [81920,256]
  prep_act_kernel<<<4096, blk256, 0, stream>>>(x, ABF, (int)M, 29, 5);
  prep_w_kernel<<<64, blk256, 0, stream>>>(g1_Wl, WB1, 256, 29, 5, 0);
  prep_w_kernel<<<64, blk256, 0, stream>>>(g1_Wr, WB2, 256, 29, 5, 0);
  {
    int blocks = (int)((M / 16) * (256 / 16) / 4);
    gemm_bf16_kernel<<<blocks, blk128, 0, stream>>>(
        ABF, WB1, 32, g1_bl, nullptr, Hl1, 256, (int)M, 256);
    gemm_bf16_kernel<<<blocks, blk128, 0, stream>>>(
        ABF, WB2, 32, g1_br, nullptr, Hr1, 256, (int)M, 256);
  }
  gat_attn_kernel<<<16384, blk256, 0, stream>>>(Hl1, Hr1, g1_att, g1_bias, edge, h1, 2);

  // ---- GATv2 layer 2: h1[81920,256] -> Hl2/Hr2 [81920,128]
  prep_act_kernel<<<4096, blk256, 0, stream>>>(h1, ABF, (int)M, 256, 8);
  prep_w_kernel<<<128, blk256, 0, stream>>>(g2_Wl, WB1, 128, 256, 8, 0);
  prep_w_kernel<<<128, blk256, 0, stream>>>(g2_Wr, WB2, 128, 256, 8, 0);
  {
    int blocks = (int)((M / 16) * (128 / 16) / 4);
    gemm_bf16_kernel<<<blocks, blk128, 0, stream>>>(
        ABF, WB1, 256, g2_bl, nullptr, Hl2, 128, (int)M, 128);
    gemm_bf16_kernel<<<blocks, blk128, 0, stream>>>(
        ABF, WB2, 256, g2_br, nullptr, Hr2, 128, (int)M, 128);
  }
  gat_attn_kernel<<<16384, blk128, 0, stream>>>(Hl2, Hr2, g2_att, g2_bias, edge, h2, 1);

  // ---- LSTM layer 0
  prep_act_kernel<<<4096, blk256, 0, stream>>>(h2, ABF, (int)M, 128, 7);
  prep_w_kernel<<<256, blk256, 0, stream>>>(Wih0, WB1, 512, 128, 7, 1);
  {
    int blocks = (int)((M / 16) * (512 / 16) / 4);
    gemm_bf16_kernel<<<blocks, blk128, 0, stream>>>(
        ABF, WB1, 128, bih0, bhh0, Xp0, 512, (int)M, 512);
  }
  lstm_recurrent_kernel<<<40, blk256, 0, stream>>>(Xp0, Whh0, out0);

  // ---- LSTM layer 1
  prep_act_kernel<<<4096, blk256, 0, stream>>>(out0, ABF, (int)M, 128, 7);
  prep_w_kernel<<<256, blk256, 0, stream>>>(Wih1, WB1, 512, 128, 7, 1);
  {
    int blocks = (int)((M / 16) * (512 / 16) / 4);
    gemm_bf16_kernel<<<blocks, blk128, 0, stream>>>(
        ABF, WB1, 128, bih1, bhh1, Xp1, 512, (int)M, 512);
  }
  lstm_recurrent_kernel<<<40, blk256, 0, stream>>>(Xp1, Whh1, outL);

  // ---- tail
  tail_kernel<<<640, blk128, 0, stream>>>(
      outL, x, ln_g, ln_b, ta_W, ta_b, sk_W1, sk_b1, sk_W2, sk_b2,
      fc_W1, fc_b1, fc_W2, fc_b2, y);
}
